// GraphVAE_58798102282548
// MI455X (gfx1250) — compile-verified, matrix operand order checked by reading, not run on hardware
//
#include <hip/hip_runtime.h>
#include <hip/hip_bf16.h>

// ---------------- CDNA5 WMMA types ----------------
typedef __attribute__((ext_vector_type(8)))  __bf16 v8bf;
typedef __attribute__((ext_vector_type(16))) __bf16 v16bf;
typedef __attribute__((ext_vector_type(8)))  float  v8f;

#define NNODE 40000
#define NEDGE 320000
#define NB    128
#define NH    4
#define DD    64

__device__ __forceinline__ unsigned short f2bf(float f) {
  unsigned u = __float_as_uint(f);
  unsigned r = u + 0x7FFFu + ((u >> 16) & 1u);   // round-to-nearest-even
  return (unsigned short)(r >> 16);
}

// fp32 [M x K] -> bf16 [M x Kp] (zero pad cols K..Kp)
__global__ void cvt_act_kernel(const float* __restrict__ X, unsigned short* __restrict__ Xb,
                               int M, int K, int Kp) {
  int idx = blockIdx.x * blockDim.x + threadIdx.x;
  if (idx >= M * Kp) return;
  int m = idx / Kp, k = idx - m * Kp;
  Xb[idx] = (k < K) ? f2bf(X[(size_t)m * K + k]) : (unsigned short)0;
}

// fp32 W [K x N] -> bf16 W^T [N x Kp] (zero pad)
__global__ void cvt_wt_kernel(const float* __restrict__ W, unsigned short* __restrict__ Wt,
                              int K, int N, int Kp) {
  int idx = blockIdx.x * blockDim.x + threadIdx.x;
  if (idx >= N * Kp) return;
  int n = idx / Kp, k = idx - n * Kp;
  Wt[idx] = (k < K) ? f2bf(W[(size_t)k * N + n]) : (unsigned short)0;
}

// assemble one 16-bit WMMA fragment (two 16B chunks per lane, per ISA layout)
__device__ __forceinline__ v16bf load_frag(const unsigned short* __restrict__ p, int k) {
  v8bf lo = *(const v8bf*)(p + k);
  v8bf hi = *(const v8bf*)(p + k + 16);
  v16bf f;
#pragma unroll
  for (int i = 0; i < 8; ++i) { f[i] = lo[i]; f[i + 8] = hi[i]; }
  return f;
}

// ---- WMMA GEMM: C[MxN] = act(A_bf16[MxKp] @ B + bias), B given as bf16 B^T [NxKp]
// one wave -> 64x32 output block: 4 M-tiles x 2 N-tiles, 8 accumulators.
// Per k-step: 6 fragment loads feed 8 v_wmma_f32_16x16x32_bf16 (register reuse 2x/4x).
// Requires M % 64 == 0, N % 32 == 0, Kp % 32 == 0 (true for all shapes here).
__global__ void wmma_gemm_kernel(const unsigned short* __restrict__ A,
                                 const unsigned short* __restrict__ Bt,
                                 const float* __restrict__ bias,
                                 float* __restrict__ C,
                                 int M, int N, int Kp, int act) {
  const int lane = threadIdx.x & 31;
  const int wave = threadIdx.x >> 5;
  const int gN = N >> 5;                       // groups of 2 N-tiles
  const int gM = M >> 6;                       // groups of 4 M-tiles
  const int gid = blockIdx.x * 8 + wave;
  if (gid >= gM * gN) return;                  // uniform per-wave exit: EXEC all-1 inside
  const int gm = gid / gN;
  const int gn = gid - gm * gN;

  const int r    = lane & 15;
  const int half = lane >> 4;                  // K-half select per 16-bit fragment layout

  const unsigned short* arow[4];
#pragma unroll
  for (int i = 0; i < 4; ++i)
    arow[i] = A + (size_t)((gm * 4 + i) * 16 + r) * Kp + half * 8;
  const unsigned short* brow[2];
#pragma unroll
  for (int j = 0; j < 2; ++j)
    brow[j] = Bt + (size_t)((gn * 2 + j) * 16 + r) * Kp + half * 8;

  const v8f vzero = {0.f, 0.f, 0.f, 0.f, 0.f, 0.f, 0.f, 0.f};
  v8f acc[4][2];
#pragma unroll
  for (int i = 0; i < 4; ++i)
#pragma unroll
    for (int j = 0; j < 2; ++j) acc[i][j] = vzero;

  for (int k = 0; k < Kp; k += 32) {
    __builtin_prefetch(arow[0] + k + 64, 0, 0);
    __builtin_prefetch(brow[0] + k + 64, 0, 0);
    v16bf av[4], bv[2];
#pragma unroll
    for (int i = 0; i < 4; ++i) av[i] = load_frag(arow[i], k);
#pragma unroll
    for (int j = 0; j < 2; ++j) bv[j] = load_frag(brow[j], k);
#pragma unroll
    for (int i = 0; i < 4; ++i)
#pragma unroll
      for (int j = 0; j < 2; ++j)
        acc[i][j] = __builtin_amdgcn_wmma_f32_16x16x32_bf16(
            false, av[i], false, bv[j], (short)0, acc[i][j], false, false);
  }

#pragma unroll
  for (int j = 0; j < 2; ++j) {
    const int col = (gn * 2 + j) * 16 + r;
    const float bb = bias ? bias[col] : 0.0f;
#pragma unroll
    for (int i = 0; i < 4; ++i) {
      const int row0 = (gm * 4 + i) * 16 + half * 8;
#pragma unroll
      for (int jj = 0; jj < 8; ++jj) {
        float vv = acc[i][j][jj] + bb;
        if (act) vv = fmaxf(vv, 0.0f);
        C[(size_t)(row0 + jj) * N + col] = vv;
      }
    }
  }
}

__global__ void fill_f32_kernel(float* __restrict__ p, float val, int n) {
  int i = blockIdx.x * blockDim.x + threadIdx.x;
  if (i < n) p[i] = val;
}

// score[e,h] = dot(q[dst,h,:], k[src,h,:]) / 8
__global__ void edge_score_kernel(const float* __restrict__ q, const float* __restrict__ k,
                                  const int* __restrict__ src, const int* __restrict__ dst,
                                  float* __restrict__ score) {
  int t = blockIdx.x * blockDim.x + threadIdx.x;
  if (t >= NEDGE * NH) return;
  int e = t >> 2, h = t & 3;
  const float4* qp = (const float4*)(q + (size_t)dst[e] * 256 + h * 64);
  const float4* kp = (const float4*)(k + (size_t)src[e] * 256 + h * 64);
  float acc = 0.f;
#pragma unroll
  for (int i = 0; i < 16; ++i) {
    float4 a = qp[i], b = kp[i];
    acc += a.x * b.x + a.y * b.y + a.z * b.z + a.w * b.w;
  }
  score[t] = acc * 0.125f;
}

// float segment-max via ordered-int atomic trick (smax pre-filled with -inf)
__global__ void seg_max_kernel(const float* __restrict__ score, const int* __restrict__ dst,
                               float* __restrict__ smax) {
  int t = blockIdx.x * blockDim.x + threadIdx.x;
  if (t >= NEDGE * NH) return;
  int e = t >> 2, h = t & 3;
  float v = score[t];
  int* p = (int*)(smax + (size_t)dst[e] * NH + h);
  if (v >= 0.f) atomicMax(p, __float_as_int(v));
  else          atomicMin((unsigned int*)p, __float_as_uint(v));
}

// ex = exp(score - smax[dst]); score <- ex; denom[dst] += ex
__global__ void edge_expsum_kernel(float* __restrict__ score, const int* __restrict__ dst,
                                   const float* __restrict__ smax, float* __restrict__ denom) {
  int t = blockIdx.x * blockDim.x + threadIdx.x;
  if (t >= NEDGE * NH) return;
  int e = t >> 2, h = t & 3;
  float ex = __expf(score[t] - smax[(size_t)dst[e] * NH + h]);
  score[t] = ex;
  atomicAdd(denom + (size_t)dst[e] * NH + h, ex);
}

// agg[dst, h, d] += (ex/denom[dst,h]) * v[src, h, d]
__global__ void edge_agg_kernel(const float* __restrict__ ex, const int* __restrict__ src,
                                const int* __restrict__ dst, const float* __restrict__ denom,
                                const float* __restrict__ v, float* __restrict__ agg) {
  long long t = (long long)blockIdx.x * blockDim.x + threadIdx.x;
  if (t >= (long long)NEDGE * 256) return;
  int e = (int)(t >> 8);
  int c = (int)(t & 255);
  int h = c >> 6;
  int d = dst[e];
  float alpha = ex[(size_t)e * NH + h] / denom[(size_t)d * NH + h];
  atomicAdd(agg + (size_t)d * 256 + c, alpha * v[(size_t)src[e] * 256 + c]);
}

// mean over heads, beta-gated skip blend, optional relu. 64 threads / node.
__global__ void gate_kernel(const float* __restrict__ agg, const float* __restrict__ skip,
                            const float* __restrict__ betaW, float* __restrict__ h,
                            int relu) {
  __shared__ float red[64];
  int n = blockIdx.x;
  int d = threadIdx.x;
  const float* a = agg + (size_t)n * 256;
  float o = 0.25f * (a[d] + a[64 + d] + a[128 + d] + a[192 + d]);
  float xr = skip[(size_t)n * 64 + d];
  red[d] = o * betaW[d] + xr * betaW[64 + d] + (o - xr) * betaW[128 + d];
  __syncthreads();
  for (int s = 32; s > 0; s >>= 1) {
    if (d < s) red[d] += red[d + s];
    __syncthreads();
  }
  float beta = 1.f / (1.f + __expf(-red[0]));
  float val = beta * xr + (1.f - beta) * o;
  if (relu) val = fmaxf(val, 0.f);
  h[(size_t)n * 64 + d] = val;
}

// column sums / sumsq over N rows (stats pre-zeroed, 128 floats)
__global__ void bn_stats_kernel(const float* __restrict__ h, float* __restrict__ stats) {
  int d = threadIdx.x;
  float s = 0.f, s2 = 0.f;
  for (int n = blockIdx.x; n < NNODE; n += gridDim.x) {
    float v = h[(size_t)n * 64 + d];
    s += v; s2 += v * v;
  }
  atomicAdd(&stats[d], s);
  atomicAdd(&stats[64 + d], s2);
}

__global__ void bn_apply_kernel(float* __restrict__ h, const float* __restrict__ stats,
                                const float* __restrict__ g, const float* __restrict__ b) {
  int t = blockIdx.x * blockDim.x + threadIdx.x;
  if (t >= NNODE * 64) return;
  int d = t & 63;
  float m   = stats[d] * (1.0f / NNODE);
  float var = stats[64 + d] * (1.0f / NNODE) - m * m;
  h[t] = (h[t] - m) * rsqrtf(var + 1e-5f) * g[d] + b[d];
}

// zg = eps * exp(logvar) + mu   (first 128 rows only)
__global__ void zlatent_kernel(const float* __restrict__ eps, const float* __restrict__ mu,
                               const float* __restrict__ logvar, float* __restrict__ zg) {
  int t = blockIdx.x * blockDim.x + threadIdx.x;
  if (t >= NB * 64) return;
  zg[t] = eps[t] * __expf(logvar[t]) + mu[t];
}

// ---------------- host side ----------------
static inline unsigned cdiv(long long a, long long b) { return (unsigned)((a + b - 1) / b); }

extern "C" void kernel_launch(void* const* d_in, const int* in_sizes, int n_in,
                              void* d_out, int out_size, void* d_ws, size_t ws_size,
                              hipStream_t stream) {
  (void)in_sizes; (void)n_in; (void)out_size; (void)ws_size;

  const float* x   = (const float*)d_in[0];
  const int*   src = (const int*)d_in[1];
  const int*   dst = src + NEDGE;
  const float* eps = (const float*)d_in[3];
  float* out = (float*)d_out;

  // d_in param indices (flattened setup_inputs order)
  const int CONV_BASE[4] = {5, 14, 23, 32};       // q.W,q.b,k.W,k.b,v.W,v.b,skip.W,skip.b,beta.W
  const int BN_BASE[3]   = {41, 43, 45};          // g,b
  const int MU_W = 47, MU_B = 48, LV_W = 49, LV_B = 50;
  const int L1_W = 51, L1_B = 52, L2_W = 53, L2_B = 54;
  const int ND_W = 55, ND_B = 56, ED_W = 57, ED_B = 58, FD_W = 59, FD_B = 60;

  // output offsets: (triu, node, feat, mu, logvar)
  const size_t OFF_TRIU = 0;
  const size_t OFF_NODE = OFF_TRIU + (size_t)NB * 10080;
  const size_t OFF_FEAT = OFF_NODE + (size_t)NB * 640;
  const size_t OFF_MU   = OFF_FEAT + (size_t)NB * 2688;
  const size_t OFF_LV   = OFF_MU   + (size_t)NNODE * 64;

  // ---- workspace carving (256B aligned) ----
  char* wp = (char*)d_ws;
  auto carve = [&](size_t bytes) -> void* {
    void* r = (void*)wp;
    wp += (bytes + 255) & ~(size_t)255;
    return r;
  };
  unsigned short* Xb  = (unsigned short*)carve((size_t)NNODE * 64 * 2);
  unsigned short* Wt  = (unsigned short*)carve((size_t)10080 * 512 * 2);
  float* qb   = (float*)carve((size_t)NNODE * 256 * 4);
  float* kb   = (float*)carve((size_t)NNODE * 256 * 4);
  float* vb   = (float*)carve((size_t)NNODE * 256 * 4);
  float* skpb = (float*)carve((size_t)NNODE * 64 * 4);
  float* agg  = (float*)carve((size_t)NNODE * 256 * 4);
  float* scr  = (float*)carve((size_t)NEDGE * NH * 4);
  float* smax = (float*)carve((size_t)NNODE * NH * 4);
  float* den  = (float*)carve((size_t)NNODE * NH * 4);
  float* hbuf = (float*)carve((size_t)NNODE * 64 * 4);
  float* stats= (float*)carve(128 * 4);
  float* zg   = (float*)carve((size_t)NB * 64 * 4);
  float* dec1 = (float*)carve((size_t)NB * 512 * 4);
  float* dec2 = (float*)carve((size_t)NB * 512 * 4);
  unsigned short* Db = (unsigned short*)carve((size_t)NB * 512 * 2);

  auto cvt_act = [&](const float* X, unsigned short* XB, int M, int K, int Kp) {
    cvt_act_kernel<<<cdiv((long long)M * Kp, 256), 256, 0, stream>>>(X, XB, M, K, Kp);
  };
  auto cvt_wt = [&](const float* W, int K, int N, int Kp) {
    cvt_wt_kernel<<<cdiv((long long)N * Kp, 256), 256, 0, stream>>>(W, Wt, K, N, Kp);
  };
  auto gemm = [&](const unsigned short* A, const float* bias, float* C,
                  int M, int N, int Kp, int act) {
    unsigned blocks = cdiv((long long)(M / 64) * (N / 32), 8);   // 64x32 per wave, 8 waves/block
    wmma_gemm_kernel<<<blocks, 256, 0, stream>>>(A, Wt, bias, C, M, N, Kp, act);
  };
  auto fill = [&](float* p, float v, long long n) {
    fill_f32_kernel<<<cdiv(n, 256), 256, 0, stream>>>(p, v, (int)n);
  };

  // ---- one transformer conv layer ----
  auto conv_layer = [&](const float* xin, int K, int pbase, float* hout, int relu) {
    int Kp = (K + 31) & ~31;                       // 42 -> 64, 64 -> 64
    cvt_act(xin, Xb, NNODE, K, Kp);
    cvt_wt((const float*)d_in[pbase + 0], K, 256, Kp);
    gemm(Xb, (const float*)d_in[pbase + 1], qb, NNODE, 256, Kp, 0);
    cvt_wt((const float*)d_in[pbase + 2], K, 256, Kp);
    gemm(Xb, (const float*)d_in[pbase + 3], kb, NNODE, 256, Kp, 0);
    cvt_wt((const float*)d_in[pbase + 4], K, 256, Kp);
    gemm(Xb, (const float*)d_in[pbase + 5], vb, NNODE, 256, Kp, 0);
    cvt_wt((const float*)d_in[pbase + 6], K, 64, Kp);
    gemm(Xb, (const float*)d_in[pbase + 7], skpb, NNODE, 64, Kp, 0);

    fill(smax, -__builtin_huge_valf(), (long long)NNODE * NH);
    fill(den, 0.f, (long long)NNODE * NH);
    fill(agg, 0.f, (long long)NNODE * 256);

    unsigned eg = cdiv((long long)NEDGE * NH, 256);
    edge_score_kernel<<<eg, 256, 0, stream>>>(qb, kb, src, dst, scr);
    seg_max_kernel<<<eg, 256, 0, stream>>>(scr, dst, smax);
    edge_expsum_kernel<<<eg, 256, 0, stream>>>(scr, dst, smax, den);
    edge_agg_kernel<<<cdiv((long long)NEDGE * 256, 256), 256, 0, stream>>>(
        scr, src, dst, den, vb, agg);
    gate_kernel<<<NNODE, 64, 0, stream>>>(agg, skpb, (const float*)d_in[pbase + 8], hout, relu);
  };

  auto batch_norm = [&](float* h, int bnbase) {
    fill(stats, 0.f, 128);
    bn_stats_kernel<<<256, 64, 0, stream>>>(h, stats);
    bn_apply_kernel<<<cdiv((long long)NNODE * 64, 256), 256, 0, stream>>>(
        h, stats, (const float*)d_in[bnbase], (const float*)d_in[bnbase + 1]);
  };

  // ---- encoder ----
  conv_layer(x, 42, CONV_BASE[0], hbuf, 1);  batch_norm(hbuf, BN_BASE[0]);
  conv_layer(hbuf, 64, CONV_BASE[1], hbuf, 1); batch_norm(hbuf, BN_BASE[1]);
  conv_layer(hbuf, 64, CONV_BASE[2], hbuf, 1); batch_norm(hbuf, BN_BASE[2]);
  conv_layer(hbuf, 64, CONV_BASE[3], hbuf, 1);  // relu, no bn

  // ---- latent heads (write straight into d_out) ----
  cvt_act(hbuf, Xb, NNODE, 64, 64);
  cvt_wt((const float*)d_in[MU_W], 64, 64, 64);
  gemm(Xb, (const float*)d_in[MU_B], out + OFF_MU, NNODE, 64, 64, 0);
  cvt_wt((const float*)d_in[LV_W], 64, 64, 64);
  gemm(Xb, (const float*)d_in[LV_B], out + OFF_LV, NNODE, 64, 64, 0);

  zlatent_kernel<<<cdiv(NB * 64, 256), 256, 0, stream>>>(eps, out + OFF_MU, out + OFF_LV, zg);

  // ---- decoder MLP (M = 128) ----
  cvt_act(zg, Db, NB, 64, 64);
  cvt_wt((const float*)d_in[L1_W], 64, 512, 64);
  gemm(Db, (const float*)d_in[L1_B], dec1, NB, 512, 64, 1);

  cvt_act(dec1, Db, NB, 512, 512);
  cvt_wt((const float*)d_in[L2_W], 512, 512, 512);
  gemm(Db, (const float*)d_in[L2_B], dec2, NB, 512, 512, 1);

  cvt_act(dec2, Db, NB, 512, 512);
  cvt_wt((const float*)d_in[ED_W], 512, 10080, 512);
  gemm(Db, (const float*)d_in[ED_B], out + OFF_TRIU, NB, 10080, 512, 0);
  cvt_wt((const float*)d_in[ND_W], 512, 640, 512);
  gemm(Db, (const float*)d_in[ND_B], out + OFF_NODE, NB, 640, 512, 0);
  cvt_wt((const float*)d_in[FD_W], 512, 2688, 512);
  gemm(Db, (const float*)d_in[FD_B], out + OFF_FEAT, NB, 2688, 512, 0);
}
